// Block_7696581394709
// MI455X (gfx1250) — compile-verified, matrix operand order checked by reading, not run on hardware
//
#include <hip/hip_runtime.h>

// ---- problem constants (B=4, I=1024, D=1024, H=16, E=2048, DH=128) ----
#define SEQ    1024
#define DM     1024
#define NH     16
#define EE     2048
#define DHD    128
#define NMRG   8224          // 4*E + 2*H
#define ROWS   4096          // B * SEQ
#define NEGV   -1e10f
#define INV_SQRT_DH 0.08838834764831845f  // 1/sqrt(128)

typedef __attribute__((ext_vector_type(16))) __bf16 v16bf;
typedef __attribute__((ext_vector_type(8)))  __bf16 v8bf;
typedef __attribute__((ext_vector_type(8)))  float  v8f;

// gcc-vector int4 in explicit address spaces, matching the async builtin's
// prototype: (v4i __device__* src, v4i __shared__* dst, imm offset, imm cpol)
typedef int vsi4 __attribute__((vector_size(16)));
typedef __attribute__((address_space(1))) vsi4 gv4i;
typedef __attribute__((address_space(3))) vsi4 lv4i;

#if defined(__AMDGCN__) && __has_builtin(__builtin_amdgcn_global_load_async_to_lds_b128) && \
    __has_builtin(__builtin_amdgcn_s_wait_asynccnt)
#define HAVE_ASYNC_LDS 1
#else
#define HAVE_ASYNC_LDS 0
#endif

// Load one WMMA 16-bit operand register for "row r over K" striping:
// elements 0..7  <- p[8*hi + 0..7], elements 8..15 <- p[16 + 8*hi + 0..7].
// Used for A (r = M-row) and, from a pre-transposed matrix, for B (r = N-col).
// Works for both global and LDS pointers (compiler picks b128 loads).
static __device__ __forceinline__ v16bf load_op16(const __bf16* p, int hi) {
  v8bf lo = *(const v8bf*)(p + 8 * hi);
  v8bf up = *(const v8bf*)(p + 16 + 8 * hi);
  v16bf r;
#pragma unroll
  for (int e = 0; e < 8; ++e) { r[e] = lo[e]; r[8 + e] = up[e]; }
  return r;
}

static __device__ __forceinline__ v8f wmma_bf16(v16bf a, v16bf b, v8f c) {
  return __builtin_amdgcn_wmma_f32_16x16x32_bf16(false, a, false, b, (short)0, c, false, false);
}

static __device__ __forceinline__ v8f zero8() {
  v8f z = {0.f, 0.f, 0.f, 0.f, 0.f, 0.f, 0.f, 0.f};
  return z;
}

// ---------------- LayerNorm (row per block) ----------------
__global__ void ln_to_bf16(const float* __restrict__ x, const float* __restrict__ g,
                           const float* __restrict__ b, __bf16* __restrict__ out) {
  const int row = blockIdx.x;
  const float* xr = x + (size_t)row * DM;
  float s = 0.f, ss = 0.f;
  for (int c = threadIdx.x; c < DM; c += blockDim.x) { float v = xr[c]; s += v; ss += v * v; }
#pragma unroll
  for (int o = 16; o > 0; o >>= 1) { s += __shfl_xor(s, o, 32); ss += __shfl_xor(ss, o, 32); }
  __shared__ float sh0[8], sh1[8];
  __shared__ float mu_s, inv_s;
  const int w = threadIdx.x >> 5, l = threadIdx.x & 31;
  if (l == 0) { sh0[w] = s; sh1[w] = ss; }
  __syncthreads();
  if (threadIdx.x == 0) {
    float a = 0.f, q = 0.f;
    for (int i = 0; i < (int)(blockDim.x >> 5); ++i) { a += sh0[i]; q += sh1[i]; }
    float mu = a / (float)DM;
    float var = q / (float)DM - mu * mu;
    mu_s = mu; inv_s = rsqrtf(var + 1e-5f);
  }
  __syncthreads();
  const float mu = mu_s, inv = inv_s;
  __bf16* orow = out + (size_t)row * DM;
  for (int c = threadIdx.x; c < DM; c += blockDim.x)
    orow[c] = (__bf16)((xr[c] - mu) * inv * g[c] + b[c]);
}

__global__ void ln_to_f32(const float* __restrict__ x, const float* __restrict__ g,
                          const float* __restrict__ b, float* __restrict__ out) {
  const int row = blockIdx.x;
  const float* xr = x + (size_t)row * DM;
  float s = 0.f, ss = 0.f;
  for (int c = threadIdx.x; c < DM; c += blockDim.x) { float v = xr[c]; s += v; ss += v * v; }
#pragma unroll
  for (int o = 16; o > 0; o >>= 1) { s += __shfl_xor(s, o, 32); ss += __shfl_xor(ss, o, 32); }
  __shared__ float sh0[8], sh1[8];
  __shared__ float mu_s, inv_s;
  const int w = threadIdx.x >> 5, l = threadIdx.x & 31;
  if (l == 0) { sh0[w] = s; sh1[w] = ss; }
  __syncthreads();
  if (threadIdx.x == 0) {
    float a = 0.f, q = 0.f;
    for (int i = 0; i < (int)(blockDim.x >> 5); ++i) { a += sh0[i]; q += sh1[i]; }
    float mu = a / (float)DM;
    float var = q / (float)DM - mu * mu;
    mu_s = mu; inv_s = rsqrtf(var + 1e-5f);
  }
  __syncthreads();
  const float mu = mu_s, inv = inv_s;
  float* orow = out + (size_t)row * DM;
  for (int c = threadIdx.x; c < DM; c += blockDim.x)
    orow[c] = (xr[c] - mu) * inv * g[c] + b[c];
}

// ---------------- weight convert + transpose (f32 [K x N] -> bf16 [N x K]) ----------------
__global__ void convert_transpose(const float* __restrict__ W, __bf16* __restrict__ Wt,
                                  int K, int N) {
  const size_t total = (size_t)K * (size_t)N;
  const size_t stride = (size_t)gridDim.x * blockDim.x;
  for (size_t idx = (size_t)blockIdx.x * blockDim.x + threadIdx.x; idx < total; idx += stride) {
    int n = (int)(idx % (size_t)N);
    int k = (int)(idx / (size_t)N);
    Wt[(size_t)n * K + k] = (__bf16)W[idx];
  }
}

// ---------------- in-projection GEMM: hn[4096x1024] @ Wm[1024x8192] + bias ----------------
// Covers only the q/k/v/p columns (8192 = 256 n-tiles), so the 8-wave groups
// divide evenly and every wave is unconditionally active (no exec-divergent
// accumulator region). One block owns a 32-row M band and 8 n-tiles; the
// shared 32x32 A chunk is double-buffered in LDS via gfx1250 async copies
// (GLOBAL_LOAD_ASYNC_TO_LDS_B128 + s_wait_asynccnt), cutting A traffic 8x.
#define NGRP 8
__global__ void gemm_in(const __bf16* __restrict__ A, const __bf16* __restrict__ Bt,
                        const float* __restrict__ bias,
                        __bf16* __restrict__ Q, __bf16* __restrict__ K,
                        __bf16* __restrict__ V, __bf16* __restrict__ P) {
  const int NT = 4 * EE / 32;                   // 256 n-tiles (q,k,v,p)
  const int NBLK = NT / NGRP;                   // 32
  const int bm = blockIdx.x / NBLK;
  const int bg = blockIdx.x % NBLK;
  const int w = threadIdx.x >> 5;
  const int nt = bg * NGRP + w;
  const int m0 = bm * 32;
  const int n0 = nt * 32;
  const int lane = threadIdx.x & 31, r = lane & 15, hi = lane >> 4;

  __shared__ __bf16 Abuf[2][32][32];            // 2 x 2KB double buffer

  // staging mapping: threads 0..127 each copy 16B (one 32x32 bf16 chunk)
  const int t = threadIdx.x;
  const int srow = t >> 2;                      // 0..31
  const int scol = (t & 3) * 8;                 // element offset in row
  auto stage = [&](int buf, int k) {
    if (t < 128) {
      const __bf16* g = A + (size_t)(m0 + srow) * DM + k + scol;
#if HAVE_ASYNC_LDS
      __builtin_amdgcn_global_load_async_to_lds_b128(
          (gv4i*)g, (lv4i*)&Abuf[buf][srow][scol], 0, 0);
#else
      *(v8bf*)&Abuf[buf][srow][scol] = *(const v8bf*)g;
#endif
    }
  };

  const __bf16* b0p = Bt + (size_t)(n0 + r) * DM;
  const __bf16* b1p = Bt + (size_t)(n0 + 16 + r) * DM;
  v8f c00 = zero8(), c01 = zero8(), c10 = zero8(), c11 = zero8();

  stage(0, 0);
  const int KSTEPS = DM / 32;
  for (int kc = 0; kc < KSTEPS; ++kc) {
    const int cur = kc & 1;
#if HAVE_ASYNC_LDS
    __builtin_amdgcn_s_wait_asynccnt(0);        // issuing waves: chunk landed in LDS
#endif
    __syncthreads();                            // publish Abuf[cur] to all waves
    if (kc + 1 < KSTEPS) stage(cur ^ 1, (kc + 1) * 32);
    const int k = kc * 32;
    v16bf a0 = load_op16(&Abuf[cur][r][0], hi);          // ds_load_b128 x2
    v16bf a1 = load_op16(&Abuf[cur][r + 16][0], hi);
    v16bf b0 = load_op16(b0p + k, hi);
    v16bf b1 = load_op16(b1p + k, hi);
    c00 = wmma_bf16(a0, b0, c00);
    c01 = wmma_bf16(a0, b1, c01);
    c10 = wmma_bf16(a1, b0, c10);
    c11 = wmma_bf16(a1, b1, c11);
  }
  auto store_t = [&](const v8f& c, int mb, int nb) {
    const int n = nb + r;
    const float bv = bias[n];
#pragma unroll
    for (int v = 0; v < 8; ++v) {
      const int m = mb + v + 8 * hi;
      const float val = c[v] + bv;
      if (n < EE)            Q[(size_t)m * EE + n]            = (__bf16)val;
      else if (n < 2 * EE)   K[(size_t)m * EE + (n - EE)]     = (__bf16)val;
      else if (n < 3 * EE)   V[(size_t)m * EE + (n - 2 * EE)] = (__bf16)val;
      else                   P[(size_t)m * EE + (n - 3 * EE)] = (__bf16)val;
    }
  };
  store_t(c00, m0, n0);
  store_t(c01, m0, n0 + 16);
  store_t(c10, m0 + 16, n0);
  store_t(c11, m0 + 16, n0 + 16);
}

// ---------------- side GEMM for smear/dpos columns (4096 x 32, K=1024) ----------------
// One wave per output row; lane l owns merged column 8192+l. Tiny FLOP count.
__global__ void gemm_sd(const __bf16* __restrict__ hn, const __bf16* __restrict__ WmT,
                        const float* __restrict__ bias, float* __restrict__ SD) {
  const int m = blockIdx.x;
  const int l = threadIdx.x;     // 0..31
  const __bf16* ar = hn + (size_t)m * DM;
  const __bf16* br = WmT + (size_t)(4 * EE + l) * DM;
  float acc = 0.f;
  for (int k = 0; k < DM; ++k) acc += (float)ar[k] * (float)br[k];
  SD[(size_t)m * 32 + l] = acc + bias[4 * EE + l];
}

// ---------------- token-shift smear gate on K ----------------
__global__ void smear_k(const __bf16* __restrict__ Kin, const float* __restrict__ SD,
                        __bf16* __restrict__ Ks) {
  const size_t total = (size_t)ROWS * EE;
  const size_t stride = (size_t)gridDim.x * blockDim.x;
  for (size_t idx = (size_t)blockIdx.x * blockDim.x + threadIdx.x; idx < total; idx += stride) {
    const int m = (int)(idx / EE), c = (int)(idx % EE);
    const int h = c / DHD;
    const float s = 1.f / (1.f + __expf(-SD[(size_t)m * 32 + h]));
    const float kc = (float)Kin[idx];
    const int i = m & (SEQ - 1);
    const float kp = (i > 0) ? (float)Kin[idx - EE] : 0.f;
    Ks[idx] = (__bf16)((1.f - s) * kc + s * kp);
  }
}

// ---------------- per-(b,h) cumulative-position scan ----------------
__global__ void pos_scan(const float* __restrict__ SD, float* __restrict__ pos) {
  const int t = threadIdx.x;  // t = b*16 + h, 64 threads
  if (t >= 4 * NH) return;
  const int b = t >> 4, h = t & 15;
  float acc = 0.f;
  for (int i = 0; i < SEQ; ++i) {
    const float d = SD[(size_t)(b * SEQ + i) * 32 + 16 + h];
    acc += 1.f / (1.f + __expf(-d));
    pos[(size_t)t * SEQ + i] = acc;
  }
}

// ---------------- V transpose: [b,i,(h dh)] -> Vt[(b h), dh, i] ----------------
__global__ void v_trans(const __bf16* __restrict__ Vin, __bf16* __restrict__ Vt) {
  const size_t total = (size_t)ROWS * EE;
  const size_t stride = (size_t)gridDim.x * blockDim.x;
  for (size_t o = (size_t)blockIdx.x * blockDim.x + threadIdx.x; o < total; o += stride) {
    const int i  = (int)(o & (SEQ - 1));
    const int dh = (int)((o >> 10) & (DHD - 1));
    const int bh = (int)(o >> 17);           // b*16 + h
    const int b = bh >> 4, h = bh & 15;
    Vt[o] = Vin[(size_t)(b * SEQ + i) * EE + h * DHD + dh];
  }
}

// ---------------- fused causal attention + silu(p) gate ----------------
// One wave owns a 16-row query tile of one (b,h). Computes transposed score
// tiles S'[j,i] = K·Qᵀ so the post-softmax probabilities land lane-local in
// exactly the A-operand striping needed for the P·V WMMA (no cross-lane swap).
__global__ void attn_kernel(const __bf16* __restrict__ Q, const __bf16* __restrict__ Ks,
                            const __bf16* __restrict__ Vt, const __bf16* __restrict__ P,
                            const float* __restrict__ pos, const float* __restrict__ log_scale,
                            __bf16* __restrict__ G) {
  const int TPH = SEQ / 16;  // 64 query tiles per head
  const int wave = blockIdx.x * (blockDim.x >> 5) + (threadIdx.x >> 5);
  if (wave >= 4 * NH * TPH) return;
  const int it = wave % TPH;
  const int bh = wave / TPH;          // b*NH + h
  const int h = bh & (NH - 1);
  const int b = bh >> 4;
  const int i0 = it * 16;
  const int lane = threadIdx.x & 31, r = lane & 15, hi = lane >> 4;

  const __bf16* Qb = Q  + (size_t)(b * SEQ) * EE + h * DHD;
  const __bf16* Kb = Ks + (size_t)(b * SEQ) * EE + h * DHD;
  const __bf16* Vb = Vt + (size_t)bh * DHD * SEQ;
  const float*  pb = pos + (size_t)bh * SEQ;

  const float sc = __expf(-2.f * log_scale[h]) * INV_SQRT_DH;

  // Q rows for this tile as B-operand (columns of Qᵀ), 4 chunks over dh=128.
  v16bf qreg[4];
#pragma unroll
  for (int kc = 0; kc < 4; ++kc)
    qreg[kc] = load_op16(Qb + (size_t)(i0 + r) * EE + kc * 32, hi);

  const int   i_col = i0 + r;      // stats column owned by this lane
  const float pos_i = pb[i_col];

  float m_run = -3.0e38f, l_run = 0.f;
  v8f o[8];
#pragma unroll
  for (int nt = 0; nt < 8; ++nt) o[nt] = zero8();

  for (int jc = 0; jc <= i0; jc += 32) {
    v8f s0 = zero8(), s1 = zero8();
#pragma unroll
    for (int kc = 0; kc < 4; ++kc) {
      v16bf a0 = load_op16(Kb + (size_t)(jc + r) * EE + kc * 32, hi);
      v16bf a1 = load_op16(Kb + (size_t)(jc + 16 + r) * EE + kc * 32, hi);
      s0 = wmma_bf16(a0, qreg[kc], s0);
      s1 = wmma_bf16(a1, qreg[kc], s1);
    }
    // scale + relpos bias + causal mask; element (j = jc + v + 8*hi [+16], i = i_col)
    float mx = -3.0e38f;
#pragma unroll
    for (int v = 0; v < 8; ++v) {
      const int j0v = jc + v + 8 * hi;
      const int j1v = j0v + 16;
      const float t0 = (j0v <= i_col) ? (s0[v] * sc + (pb[j0v] - pos_i)) : NEGV;
      const float t1 = (j1v <= i_col) ? (s1[v] * sc + (pb[j1v] - pos_i)) : NEGV;
      s0[v] = t0; s1[v] = t1;
      mx = fmaxf(mx, fmaxf(t0, t1));
    }
    mx = fmaxf(mx, __shfl_xor(mx, 16, 32));
    const float m_new = fmaxf(m_run, mx);
    const float alpha = __expf(m_run - m_new);
    float rs = 0.f;
    v16bf pa;  // probabilities already in A-operand striping (lane-local)
#pragma unroll
    for (int v = 0; v < 8; ++v) {
      const float p0 = __expf(s0[v] - m_new);
      const float p1 = __expf(s1[v] - m_new);
      rs += p0 + p1;
      pa[v]     = (__bf16)p0;
      pa[8 + v] = (__bf16)p1;
    }
    rs += __shfl_xor(rs, 16, 32);
    l_run = l_run * alpha + rs;
    m_run = m_new;
    // broadcast alpha per O-row (row = v + 8*hi, stats live in lane == row)
    float arow[8];
#pragma unroll
    for (int v = 0; v < 8; ++v) arow[v] = __shfl(alpha, v + 8 * hi, 32);
#pragma unroll
    for (int nt = 0; nt < 8; ++nt) {
      v16bf bv = load_op16(Vb + (size_t)(nt * 16 + r) * SEQ + jc, hi);
      v8f oo = o[nt];
#pragma unroll
      for (int v = 0; v < 8; ++v) oo[v] *= arow[v];
      o[nt] = wmma_bf16(pa, bv, oo);
    }
  }
  float lrow[8];
#pragma unroll
  for (int v = 0; v < 8; ++v) lrow[v] = __shfl(l_run, v + 8 * hi, 32);
#pragma unroll
  for (int nt = 0; nt < 8; ++nt) {
#pragma unroll
    for (int v = 0; v < 8; ++v) {
      const int i = i0 + v + 8 * hi;
      const int n = h * DHD + nt * 16 + r;
      const size_t idx = (size_t)(b * SEQ + i) * EE + n;
      const float ov = o[nt][v] / lrow[v];
      const float pv = (float)P[idx];
      const float gate = pv / (1.f + __expf(-pv));   // silu
      G[idx] = (__bf16)(gate * ov);
    }
  }
}

// ---------------- out-projection GEMM: G[4096x2048] @ Wo[2048x1024] -> f32 ----------------
__global__ void gemm_out(const __bf16* __restrict__ A, const __bf16* __restrict__ Bt,
                         float* __restrict__ C) {
  const int NT = DM / 32;  // 32
  const int wave = blockIdx.x * (blockDim.x >> 5) + (threadIdx.x >> 5);
  if (wave >= (ROWS / 32) * NT) return;
  const int m0 = (wave / NT) * 32;
  const int n0 = (wave % NT) * 32;
  const int lane = threadIdx.x & 31, r = lane & 15, hi = lane >> 4;
  const __bf16* a0p = A + (size_t)(m0 + r) * EE;
  const __bf16* a1p = A + (size_t)(m0 + 16 + r) * EE;
  const __bf16* b0p = Bt + (size_t)(n0 + r) * EE;
  const __bf16* b1p = Bt + (size_t)(n0 + 16 + r) * EE;
  v8f c00 = zero8(), c01 = zero8(), c10 = zero8(), c11 = zero8();
  for (int k = 0; k < EE; k += 32) {
    v16bf a0 = load_op16(a0p + k, hi);
    v16bf a1 = load_op16(a1p + k, hi);
    v16bf b0 = load_op16(b0p + k, hi);
    v16bf b1 = load_op16(b1p + k, hi);
    c00 = wmma_bf16(a0, b0, c00);
    c01 = wmma_bf16(a0, b1, c01);
    c10 = wmma_bf16(a1, b0, c10);
    c11 = wmma_bf16(a1, b1, c11);
  }
  auto store_t = [&](const v8f& c, int mb, int nb) {
    const int n = nb + r;
#pragma unroll
    for (int v = 0; v < 8; ++v) {
      const int m = mb + v + 8 * hi;
      C[(size_t)m * DM + n] = c[v];
    }
  };
  store_t(c00, m0, n0);
  store_t(c01, m0, n0 + 16);
  store_t(c10, m0 + 16, n0);
  store_t(c11, m0 + 16, n0 + 16);
}

// ---------------- launch ----------------
extern "C" void kernel_launch(void* const* d_in, const int* in_sizes, int n_in,
                              void* d_out, int out_size, void* d_ws, size_t ws_size,
                              hipStream_t stream) {
  const float* x     = (const float*)d_in[0];
  const float* Wm    = (const float*)d_in[1];
  const float* bm    = (const float*)d_in[2];
  const float* ln1g  = (const float*)d_in[3];
  const float* ln1b  = (const float*)d_in[4];
  const float* logsc = (const float*)d_in[5];
  const float* Wo    = (const float*)d_in[6];
  const float* ln2g  = (const float*)d_in[7];
  const float* ln2b  = (const float*)d_in[8];
  (void)in_sizes; (void)n_in; (void)out_size; (void)ws_size;

  char* ws = (char*)d_ws;
  size_t off = 0;
  auto carve = [&](size_t bytes) -> char* {
    char* p = ws + off;
    off += (bytes + 255) & ~(size_t)255;
    return p;
  };
  __bf16* hn   = (__bf16*)carve((size_t)ROWS * DM * 2);
  __bf16* WmT  = (__bf16*)carve((size_t)NMRG * DM * 2);
  __bf16* WoT  = (__bf16*)carve((size_t)DM * EE * 2);
  __bf16* Qb   = (__bf16*)carve((size_t)ROWS * EE * 2);
  __bf16* Kb   = (__bf16*)carve((size_t)ROWS * EE * 2);
  __bf16* Ksb  = (__bf16*)carve((size_t)ROWS * EE * 2);
  __bf16* Vp   = (__bf16*)carve((size_t)ROWS * EE * 2);
  __bf16* Vt   = (__bf16*)carve((size_t)ROWS * EE * 2);
  __bf16* Pb   = (__bf16*)carve((size_t)ROWS * EE * 2);
  __bf16* Gb   = (__bf16*)carve((size_t)ROWS * EE * 2);
  float*  SD   = (float*) carve((size_t)ROWS * 32 * 4);
  float*  posb = (float*) carve((size_t)4 * NH * SEQ * 4);
  float*  Opre = (float*) carve((size_t)ROWS * DM * 4);

  ln_to_bf16<<<ROWS, 256, 0, stream>>>(x, ln1g, ln1b, hn);
  convert_transpose<<<4096, 256, 0, stream>>>(Wm, WmT, DM, NMRG);
  convert_transpose<<<1024, 256, 0, stream>>>(Wo, WoT, EE, DM);
  {
    const int NT = 4 * EE / 32;                  // 256
    const int NBLK = NT / NGRP;                  // 32
    gemm_in<<<(ROWS / 32) * NBLK, 256, 0, stream>>>(hn, WmT, bm, Qb, Kb, Vp, Pb);
  }
  gemm_sd<<<ROWS, 32, 0, stream>>>(hn, WmT, bm, SD);
  smear_k<<<4096, 256, 0, stream>>>(Kb, SD, Ksb);
  pos_scan<<<1, 64, 0, stream>>>(SD, posb);
  v_trans<<<4096, 256, 0, stream>>>(Vp, Vt);
  {
    const int waves = 4 * NH * (SEQ / 16);       // 4096
    attn_kernel<<<(waves + 7) / 8, 256, 0, stream>>>(Qb, Ksb, Vt, Pb, posb, logsc, Gb);
  }
  {
    const int waves = (ROWS / 32) * (DM / 32);   // 4096
    gemm_out<<<(waves + 7) / 8, 256, 0, stream>>>(Gb, WoT, Opre);
  }
  ln_to_f32<<<ROWS, 256, 0, stream>>>(Opre, ln2g, ln2b, (float*)d_out);
}
